// TransformerBlock_78769700209269
// MI455X (gfx1250) — matrix operand order repaired and checked by reading, not compiled, not run on hardware
//
#include <hip/hip_runtime.h>
#include <cstdint>
#include <cstddef>

// ---------------------------------------------------------------------------
// CDNA5 / gfx1250 transformer block: bf16 WMMA (f32 accum) everywhere.
// ---------------------------------------------------------------------------

typedef __attribute__((ext_vector_type(16))) __bf16       v16bf;
typedef __attribute__((ext_vector_type(8)))  float        v8f;
typedef __attribute__((ext_vector_type(8)))  unsigned int v8u;

#define WMMA_BF16(a, b, c) \
  __builtin_amdgcn_wmma_f32_16x16x32_bf16(false, (a), false, (b), (short)0, (c), false, false)

constexpr int S_  = 2048;
constexpr int B_  = 4;
constexpr int D_  = 1024;
constexpr int H_  = 2;
constexpr int F_  = 2048;
constexpr int HD_ = 512;
constexpr int NROW = S_ * B_;   // 8192 token rows

// ---------------------------------------------------------------------------
// RMSNorm(x, w) -> bf16 output.  One block per row, 256 threads.
// ---------------------------------------------------------------------------
__global__ __launch_bounds__(256) void rmsnorm_cast_kernel(
    const float* __restrict__ x, const float* __restrict__ w,
    __bf16* __restrict__ out)
{
  __shared__ float red[256];
  const int row = blockIdx.x;
  const int tid = threadIdx.x;
  const float* xr = x + (size_t)row * D_;

  float s = 0.f;
#pragma unroll
  for (int j = 0; j < 4; ++j) { float v = xr[tid + j * 256]; s += v * v; }
  red[tid] = s;
  __syncthreads();
  for (int off = 128; off > 0; off >>= 1) {
    if (tid < off) red[tid] += red[tid + off];
    __syncthreads();
  }
  const float inv = rsqrtf(red[0] / (float)D_ + 1e-8f);
  __bf16* orow = out + (size_t)row * D_;
#pragma unroll
  for (int j = 0; j < 4; ++j) {
    int i = tid + j * 256;
    orow[i] = (__bf16)(xr[i] * inv * w[i]);
  }
}

// ---------------------------------------------------------------------------
// Tiled GEMM:  C[M,N] = A[M,K](bf16,row-major) * W[N,K]^T + bias
// 128x128 output tile per 256-thread block (8 waves), BK=32.
// Software-pipelined: global->reg loads for step it+1 issue before the WMMAs
// of step it; reg->LDS stores follow the WMMAs; double-buffered LDS, one
// barrier per step; global_prefetch two steps ahead.
// Wave w: wm = w>>2 -> 64 M-rows (4 tiles), wn = w&3 -> 32 N-cols (2 tiles)
// -> 8 WMMAs / wave / k-step.  Epilogue templated.
// ---------------------------------------------------------------------------
constexpr int MODE_QKV  = 0;
constexpr int MODE_OUT  = 1;
constexpr int MODE_MLP1 = 2;
constexpr int MODE_MLP2 = 3;

template <int MODE>
__global__ __launch_bounds__(256) void gemm_kernel(
    const __bf16* __restrict__ A, const float* __restrict__ W,
    const float* __restrict__ bias, const float* __restrict__ resid,
    float* __restrict__ outF, __bf16* __restrict__ outB,
    __bf16* __restrict__ qout, __bf16* __restrict__ kout,
    __bf16* __restrict__ vout, int K)
{
  __shared__ __bf16 As[2][128][32];   // 16 KB
  __shared__ __bf16 Bs[2][128][32];   // 16 KB

  const int tid  = threadIdx.x;
  const int lane = tid & 31;
  const int wv   = tid >> 5;
  const int half = lane >> 4;
  const int l16  = lane & 15;
  const int gm0  = blockIdx.y * 128;
  const int gn0  = blockIdx.x * 128;
  const int wm   = wv >> 2;           // 0..1 : 64-row slab
  const int wn   = wv & 3;            // 0..3 : 32-col slab

  v8f acc[4][2] = {};

  // Per-thread staging: row = tid>>1 (0..127), 16 contiguous elements.
  const int ldr = tid >> 1;
  const int ldc = (tid & 1) * 16;

  uint4  areg0, areg1;   // A-tile staging registers (16 bf16)
  float4 wreg[4];        // W-tile staging registers (16 f32)

  auto gload = [&](int kk) {
    const __bf16* ap = A + (size_t)(gm0 + ldr) * K + kk + ldc;
    areg0 = *(const uint4*)ap;
    areg1 = *(const uint4*)(ap + 8);
    const float4* wp = (const float4*)(W + (size_t)(gn0 + ldr) * K + kk + ldc);
    wreg[0] = wp[0]; wreg[1] = wp[1]; wreg[2] = wp[2]; wreg[3] = wp[3];
    // Pull the k-step after this one toward the caches (global_prefetch_b8).
    if (kk + 32 < K) {
      __builtin_prefetch(A + (size_t)(gm0 + ldr) * K + kk + 32 + ldc, 0, 0);
      __builtin_prefetch(W + (size_t)(gn0 + ldr) * K + kk + 32 + ldc, 0, 0);
    }
  };
  auto lstore = [&](int buf) {
    *(uint4*)&As[buf][ldr][ldc]     = areg0;
    *(uint4*)&As[buf][ldr][ldc + 8] = areg1;
    const float* wf = (const float*)wreg;
#pragma unroll
    for (int i = 0; i < 16; ++i) Bs[buf][ldr][ldc + i] = (__bf16)wf[i];
  };

  gload(0);
  lstore(0);
  __syncthreads();

  const int nIt = K >> 5;
  for (int it = 0; it < nIt; ++it) {
    const int cur = it & 1;
    if (it + 1 < nIt) gload((it + 1) << 5);   // issue loads, no wait here

    // B fragments (2): B[k][n] = W[n][k]; lane pairs contiguous in k.
    v16bf bfr[2];
#pragma unroll
    for (int nt = 0; nt < 2; ++nt) {
      const int nrow = wn * 32 + nt * 16 + l16;
      v8u bp;
#pragma unroll
      for (int p = 0; p < 8; ++p)
        bp[p] = *(const unsigned*)&Bs[cur][nrow][half * 16 + 2 * p];
      bfr[nt] = __builtin_bit_cast(v16bf, bp);
    }
    // A fragments (4) + 8 WMMAs.
#pragma unroll
    for (int mt = 0; mt < 4; ++mt) {
      const int mrow = wm * 64 + mt * 16 + l16;
      v8u ap;
#pragma unroll
      for (int p = 0; p < 8; ++p) {
        const int kb = half * 16 + 2 * p;   // even
        ap[p] = *(const unsigned*)&As[cur][mrow][kb];
      }
      const v16bf af = __builtin_bit_cast(v16bf, ap);
#pragma unroll
      for (int nt = 0; nt < 2; ++nt)
        acc[mt][nt] = WMMA_BF16(af, bfr[nt], acc[mt][nt]);
    }

    if (it + 1 < nIt) lstore(cur ^ 1);        // loadcnt wait lands here
    __syncthreads();
  }

  // Epilogue.  D layout: row = r + 8*half, col = lane%16.
#pragma unroll
  for (int mt = 0; mt < 4; ++mt) {
#pragma unroll
    for (int nt = 0; nt < 2; ++nt) {
#pragma unroll
      for (int r = 0; r < 8; ++r) {
        const int gm = gm0 + wm * 64 + mt * 16 + r + 8 * half;
        const int gn = gn0 + wn * 32 + nt * 16 + l16;
        float v = acc[mt][nt][r] + bias[gn];

        if constexpr (MODE == MODE_QKV) {
          // qkv col: [0,1024)=q [1024,2048)=k [2048,3072)=v ; head = within/512
          const int sec    = gn >> 10;
          const int within = gn & 1023;
          const int hh     = within >> 9;
          const int hd     = within & 511;
          const int srow   = gm >> 2;        // row = s*B + b
          const int b      = gm & 3;
          __bf16* dst = (sec == 0) ? qout : (sec == 1 ? kout : vout);
          dst[(((size_t)b * H_ + hh) * S_ + srow) * HD_ + hd] = (__bf16)v;
        } else if constexpr (MODE == MODE_OUT) {
          const size_t idx = (size_t)gm * D_ + gn;
          outF[idx] = v + resid[idx];
        } else if constexpr (MODE == MODE_MLP1) {
          const float g = 0.5f * v * (1.f + erff(v * 0.70710678118654752f));
          outB[(size_t)gm * F_ + gn] = (__bf16)g;
        } else {  // MODE_MLP2
          const size_t idx = (size_t)gm * D_ + gn;
          outF[idx] = v + resid[idx];
        }
      }
    }
  }
}

// ---------------------------------------------------------------------------
// Flash attention, one (b,h) per blockIdx.y, 32 queries per block.
// 256 threads = 8 waves.  Scores: wave w -> tile (mt=w&1, nt=w>>1) of the
// 32x64 score block.  Both Q and K fragments are gathered DIRECTLY from
// global memory (row-major K rows match the B-fragment k-pair layout), so
// the scores loop has no LDS traffic and no barriers -- the compiler can
// keep many chunks' loads in flight against the WMMAs.  K rows are read by
// 2 waves each and stay L2-resident (2 MB/head vs 192 MB L2).
// V goes through LDS (k-strided fragments need transposition), pipelined.
// Ctx: wave w owns hd columns [w*64, w*64+64).
// ---------------------------------------------------------------------------
__global__ __launch_bounds__(256) void attention_kernel(
    const __bf16* __restrict__ qb, const __bf16* __restrict__ kb,
    const __bf16* __restrict__ vb, __bf16* __restrict__ ctx)
{
  __shared__ __bf16 Vs[32][512];     // 32 KB  (key chunk x hd, row-major)
  __shared__ float  Ss[32][64];      //  8 KB
  __shared__ __bf16 Ps[32][64];      //  4 KB
  __shared__ float  rmax[32], rsum[32], ralpha[32];

  const int tid  = threadIdx.x;
  const int lane = tid & 31;
  const int wv   = tid >> 5;
  const int half = lane >> 4;
  const int l16  = lane & 15;
  const int bh   = blockIdx.y;
  const int qbase = blockIdx.x * 32;
  const size_t base = (size_t)bh * S_ * HD_;
  const float scale = 0.044194173824159216f;  // 1/sqrt(512)

  if (tid < 32) { rmax[tid] = -1e30f; rsum[tid] = 0.f; }
  __syncthreads();

  v8f acc[2][4] = {};
  const int smt = wv & 1;
  const int snt = wv >> 1;
  const int numKT = ((qbase + 31) >> 6) + 1;  // causal: last key tile touched

  // Per-lane global row pointers (dword-indexed) for Q / K fragment gathers.
  const unsigned* qrow =
      (const unsigned*)(qb + base + (size_t)(qbase + smt * 16 + l16) * HD_);

  for (int kt = 0; kt < numKT; ++kt) {
    const int ktbase = kt * 64;

    uint4 vreg[8];
    auto vload = [&](int kc) {
      const int keyl = tid >> 3, c0 = (tid & 7) * 64;
      const __bf16* vrow = vb + base + (size_t)(ktbase + kc + keyl) * HD_ + c0;
#pragma unroll
      for (int i = 0; i < 8; ++i) vreg[i] = *(const uint4*)(vrow + i * 8);
    };
    auto vstore = [&]() {
      const int keyl = tid >> 3, c0 = (tid & 7) * 64;
#pragma unroll
      for (int i = 0; i < 8; ++i) *(uint4*)&Vs[keyl][c0 + i * 8] = vreg[i];
    };

    // ---- scores = Q @ K^T over kdim=512: pure global gathers + WMMA --------
    const unsigned* krow =
        (const unsigned*)(kb + base + (size_t)(ktbase + snt * 16 + l16) * HD_);
    v8f sacc = {};
#pragma unroll
    for (int ix = 0; ix < (HD_ >> 5); ++ix) {
      const int kd = ix << 4;      // dword offset of this 32-wide k-chunk
      v8u ap, bp;
#pragma unroll
      for (int p = 0; p < 8; ++p) {
        const int kbp = (p < 4 ? 2 * p : 2 * p + 8) + half * 8;   // even
        ap[p] = qrow[kd + half * 8 + p];
        bp[p] = krow[kd + half * 8 + p];
      }
      const v16bf af = __builtin_bit_cast(v16bf, ap);
      const v16bf bf = __builtin_bit_cast(v16bf, bp);
      sacc = WMMA_BF16(af, bf, sacc);
    }

    // ---- scale + causal mask -> Ss ----------------------------------------
#pragma unroll
    for (int r = 0; r < 8; ++r) {
      const int m = smt * 16 + r + half * 8;
      const int n = snt * 16 + l16;
      float v = sacc[r] * scale;
      if (ktbase + n > qbase + m) v = -1e30f;
      Ss[m][n] = v;
    }
    vload(0);          // issue V chunk-0 loads; latency hides behind softmax
    __syncthreads();

    // ---- online softmax (one thread per query row) ------------------------
    if (tid < 32) {
      const int r = tid;
      float mx = -1e30f;
      for (int j = 0; j < 64; ++j) mx = fmaxf(mx, Ss[r][j]);
      const float mold = rmax[r];
      const float mnew = fmaxf(mold, mx);
      const float a    = __expf(mold - mnew);
      float s = 0.f;
      for (int j = 0; j < 64; ++j) {
        const float p = __expf(Ss[r][j] - mnew);
        Ps[r][j] = (__bf16)p;
        s += p;
      }
      rsum[r]   = rsum[r] * a + s;
      rmax[r]   = mnew;
      ralpha[r] = a;
    }
    __syncthreads();

    // ---- rescale running ctx accumulators ---------------------------------
#pragma unroll
    for (int cm = 0; cm < 2; ++cm)
#pragma unroll
      for (int cn = 0; cn < 4; ++cn)
#pragma unroll
        for (int r = 0; r < 8; ++r) {
          const int m = cm * 16 + r + half * 8;
          acc[cm][cn][r] *= ralpha[m];
        }

    // ---- ctx += P @ V, key chunks of 32 (single LDS buffer, pipelined) -----
    vstore();
    __syncthreads();
    for (int kci = 0; kci < 2; ++kci) {
      const int kc = kci * 32;
      if (kci == 0) vload(32);                 // issue chunk-1 loads

      // B fragments for V (4, hoisted): B[k][n] = Vs[k][col], k = half*16+j.
      v16bf vf[4];
#pragma unroll
      for (int cn = 0; cn < 4; ++cn) {
        const int col = wv * 64 + cn * 16 + l16;
#pragma unroll
        for (int j = 0; j < 16; ++j) vf[cn][j] = Vs[half * 16 + j][col];
      }

#pragma unroll
      for (int cm = 0; cm < 2; ++cm) {
        const int prow = cm * 16 + l16;
        v8u ap2;
#pragma unroll
        for (int p = 0; p < 8; ++p) {
          const int kbp = half * 16 + 2 * p;
          ap2[p] = *(const unsigned*)&Ps[prow][kc + kbp];
        }
        const v16bf pf = __builtin_bit_cast(v16bf, ap2);
#pragma unroll
        for (int cn = 0; cn < 4; ++cn)
          acc[cm][cn] = WMMA_BF16(pf, vf[cn], acc[cm][cn]);
      }

      if (kci == 0) {
        __syncthreads();   // all waves done reading chunk 0
        vstore();          // loadcnt wait lands here, after chunk-0 WMMAs
      }
      __syncthreads();
    }
  }

  // ---- finalize: ctx /= rsum, scatter to (s*B+b, h*HD+hd) as bf16 ----------
  const int b  = bh >> 1;
  const int hh = bh & 1;
#pragma unroll
  for (int cm = 0; cm < 2; ++cm)
#pragma unroll
    for (int cn = 0; cn < 4; ++cn)
#pragma unroll
      for (int r = 0; r < 8; ++r) {
        const int m    = cm * 16 + r + half * 8;
        const int srow = qbase + m;
        const int col  = wv * 64 + cn * 16 + l16;
        const float v  = acc[cm][cn][r] / rsum[m];
        ctx[((size_t)srow * B_ + b) * D_ + hh * HD_ + col] = (__bf16)v;
      }
}

// ---------------------------------------------------------------------------
// Host launcher
// ---------------------------------------------------------------------------
extern "C" void kernel_launch(void* const* d_in, const int* in_sizes, int n_in,
                              void* d_out, int out_size, void* d_ws,
                              size_t ws_size, hipStream_t stream)
{
  (void)in_sizes; (void)n_in; (void)out_size; (void)ws_size;

  const float* x       = (const float*)d_in[0];
  const float* w_norm1 = (const float*)d_in[1];
  const float* in_w    = (const float*)d_in[2];
  const float* in_b    = (const float*)d_in[3];
  const float* out_w   = (const float*)d_in[4];
  const float* out_b   = (const float*)d_in[5];
  const float* w_norm2 = (const float*)d_in[6];
  const float* W1      = (const float*)d_in[7];
  const float* b1      = (const float*)d_in[8];
  const float* W2      = (const float*)d_in[9];
  const float* b2      = (const float*)d_in[10];
  float* out = (float*)d_out;

  char* ws = (char*)d_ws;
  size_t off = 0;
  auto alloc = [&](size_t bytes) -> char* {
    char* p = ws + off;
    off += (bytes + 255) & ~(size_t)255;
    return p;
  };
  __bf16* h   = (__bf16*)alloc((size_t)NROW * D_ * 2);          // 16 MB
  __bf16* q   = (__bf16*)alloc((size_t)B_ * H_ * S_ * HD_ * 2); // 16 MB
  __bf16* k   = (__bf16*)alloc((size_t)B_ * H_ * S_ * HD_ * 2); // 16 MB
  __bf16* v   = (__bf16*)alloc((size_t)B_ * H_ * S_ * HD_ * 2); // 16 MB
  __bf16* cx  = (__bf16*)alloc((size_t)NROW * D_ * 2);          // 16 MB
  float*  x1  = (float*) alloc((size_t)NROW * D_ * 4);          // 32 MB
  __bf16* mb  = (__bf16*)alloc((size_t)NROW * D_ * 2);          // 16 MB
  __bf16* t1  = (__bf16*)alloc((size_t)NROW * F_ * 2);          // 32 MB

  const dim3 blk(256);

  // 1) h = rmsnorm(x, w_norm1)  (bf16)
  rmsnorm_cast_kernel<<<NROW, blk, 0, stream>>>(x, w_norm1, h);

  // 2) qkv = h @ in_proj_w^T + b -> scatter to q/k/v head-major bf16
  gemm_kernel<MODE_QKV><<<dim3(3 * D_ / 128, NROW / 128), blk, 0, stream>>>(
      h, in_w, in_b, nullptr, nullptr, nullptr, q, k, v, D_);

  // 3) flash attention -> ctx (bf16, token-row layout)
  attention_kernel<<<dim3(S_ / 32, B_ * H_), blk, 0, stream>>>(q, k, v, cx);

  // 4) x1 = x + ctx @ out_proj_w^T + out_proj_b  (f32)
  gemm_kernel<MODE_OUT><<<dim3(D_ / 128, NROW / 128), blk, 0, stream>>>(
      cx, out_w, out_b, x, x1, nullptr, nullptr, nullptr, nullptr, D_);

  // 5) m = rmsnorm(x1, w_norm2) (bf16)
  rmsnorm_cast_kernel<<<NROW, blk, 0, stream>>>(x1, w_norm2, mb);

  // 6) t1 = gelu(m @ W1^T + b1) (bf16)
  gemm_kernel<MODE_MLP1><<<dim3(F_ / 128, NROW / 128), blk, 0, stream>>>(
      mb, W1, b1, nullptr, nullptr, t1, nullptr, nullptr, nullptr, D_);

  // 7) out = x1 + t1 @ W2^T + b2 (f32)
  gemm_kernel<MODE_MLP2><<<dim3(D_ / 128, NROW / 128), blk, 0, stream>>>(
      t1, W2, b2, x1, out, nullptr, nullptr, nullptr, nullptr, F_);
}
